// GAT_64965675320025
// MI455X (gfx1250) — matrix operand order repaired and checked by reading, not compiled, not run on hardware
//
#include <hip/hip_runtime.h>
#include <hip/hip_bf16.h>

// ---------------- problem constants (match reference) ----------------
#define NN      40000     // nodes
#define EE      640000    // edges
#define IN_DIM  256
#define HID     64
#define HEADS   4
#define CLS     40
#define NEG_SLOPE 0.2f

typedef _Float16 v8h  __attribute__((ext_vector_type(8)));
typedef _Float16 v16h __attribute__((ext_vector_type(16)));
typedef float    v8f  __attribute__((ext_vector_type(8)));

union frag16 { v16h v; v8h h[2]; };
union fragC  { v8f  v; float f[8]; };

// ---------------- helpers ----------------
__global__ void fill_f32(float* __restrict__ p, float v, int n) {
    int i = blockIdx.x * blockDim.x + threadIdx.x;
    if (i < n) p[i] = v;
}

__global__ void cvt_f16(const float* __restrict__ in, _Float16* __restrict__ out, int n) {
    int i = blockIdx.x * blockDim.x + threadIdx.x;
    if (i < n) out[i] = (_Float16)in[i];
}

// Wt[f, k] = W[k, f]  (fp32 -> fp16, transposed so B fragments load contiguously)
__global__ void cvt_w_t(const float* __restrict__ W, _Float16* __restrict__ Wt,
                        int K, int F) {
    int i = blockIdx.x * blockDim.x + threadIdx.x;
    if (i >= K * F) return;
    int f = i / K, k = i % K;
    Wt[(size_t)f * K + k] = (_Float16)W[(size_t)k * F + f];
}

// ---------------- WMMA GEMM: C[M,F] = A[M,K] * Wt[F,K]^T ----------------
// One wave per 16x16 output tile, K stepped by 32 via v_wmma_f32_16x16x32_f16.
// A fragment layout (16-bit A 16x32, ISA 7.12.2): lane = (hs<<4)|m,
//   elems 0..7 -> K = k0 + 8*hs + e ; elems 8..15 -> K = k0 + 16 + 8*hs + (e-8)
// B fragment (32x16): lane = (hs<<4)|n, elems 0..7 -> K = k0 + 8*hs + e ; elems 8..15 -> K = k0 + 16 + 8*hs + (e-8)
// C/D (f32 16x16): lane = (hs<<4)|n, VGPR r -> row M = 8*hs + r.
__global__ void gemm_wmma_f16(const _Float16* __restrict__ A,
                              const _Float16* __restrict__ Bt,
                              float* __restrict__ C,
                              int K, int F) {
    const int lane = threadIdx.x & 31;
    const int wave = threadIdx.x >> 5;
    const int tm   = blockIdx.x * 4 + wave;   // M tile (M = 40000 = 625*4*16)
    const int tn   = blockIdx.y;              // N tile
    const int lo16 = lane & 15;
    const int hs   = lane >> 4;

    const _Float16* arow = A  + (size_t)(tm * 16 + lo16) * K;
    const _Float16* brow = Bt + (size_t)(tn * 16 + lo16) * K;

    v8f acc = {};
    for (int k0 = 0; k0 < K; k0 += 32) {
        frag16 a, b;
        a.h[0] = *(const v8h*)(arow + k0 + 8 * hs);
        a.h[1] = *(const v8h*)(arow + k0 + 16 + 8 * hs);
        b.h[0] = *(const v8h*)(brow + k0 + 8 * hs);
        b.h[1] = *(const v8h*)(brow + k0 + 16 + 8 * hs);
        acc = __builtin_amdgcn_wmma_f32_16x16x32_f16(
                  false, a.v, false, b.v, (short)0, acc, false, false);
    }

    fragC c; c.v = acc;
    float* crow = C + (size_t)(tm * 16 + 8 * hs) * F + tn * 16 + lo16;
    #pragma unroll
    for (int r = 0; r < 8; ++r) crow[(size_t)r * F] = c.f[r];
}

// ---------------- attention coefficients ----------------
// el[n,h] = dot(feat[n,h,:], al[h,:]) ; er likewise
__global__ void attn_coef(const float* __restrict__ feat,
                          const float* __restrict__ al,
                          const float* __restrict__ ar,
                          float* __restrict__ el, float* __restrict__ er,
                          int D) {
    int i = blockIdx.x * blockDim.x + threadIdx.x;      // i = n*HEADS + h
    if (i >= NN * HEADS) return;
    int h = i % HEADS, n = i / HEADS;
    const float* f = feat + (size_t)n * HEADS * D + h * D;
    const float* pa = al + h * D;
    const float* pb = ar + h * D;
    float sl = 0.f, sr = 0.f;
    for (int d = 0; d < D; ++d) { float v = f[d]; sl += v * pa[d]; sr += v * pb[d]; }
    el[i] = sl; er[i] = sr;
}

__device__ __forceinline__ float leaky(float x) {
    return x > 0.f ? x : NEG_SLOPE * x;
}

__device__ __forceinline__ void atomicMaxF(float* addr, float val) {
    int* ai = (int*)addr;
    int old = __float_as_int(*addr);
    while (__int_as_float(old) < val) {
        int assumed = old;
        old = atomicCAS(ai, assumed, __float_as_int(val));
        if (old == assumed) break;
    }
}

// pass 1: segment max over dst
__global__ void edge_max(const int* __restrict__ src, const int* __restrict__ dst,
                         const float* __restrict__ el, const float* __restrict__ er,
                         float* __restrict__ m) {
    int i = blockIdx.x * blockDim.x + threadIdx.x;      // i = eid*HEADS + h
    if (i >= EE * HEADS) return;
    int h = i % HEADS, eid = i / HEADS;
    float e = leaky(el[src[eid] * HEADS + h] + er[dst[eid] * HEADS + h]);
    atomicMaxF(&m[dst[eid] * HEADS + h], e);
}

// pass 2: w = exp(e - m[dst]); s[dst] += w
__global__ void edge_expsum(const int* __restrict__ src, const int* __restrict__ dst,
                            const float* __restrict__ el, const float* __restrict__ er,
                            const float* __restrict__ m,
                            float* __restrict__ wbuf, float* __restrict__ s) {
    int i = blockIdx.x * blockDim.x + threadIdx.x;
    if (i >= EE * HEADS) return;
    int h = i % HEADS, eid = i / HEADS;
    int dh = dst[eid] * HEADS + h;
    float e = leaky(el[src[eid] * HEADS + h] + er[dh]);
    float w = __expf(e - m[dh]);
    wbuf[i] = w;
    atomicAdd(&s[dh], w);
}

// pass 3: agg[dst, h, d] += (w / s[dst,h]) * feat[src, h, d]
// one block per edge, blockDim.x == HEADS*D
__global__ void edge_aggregate(const int* __restrict__ src, const int* __restrict__ dst,
                               const float* __restrict__ wbuf, const float* __restrict__ s,
                               const float* __restrict__ feat, float* __restrict__ agg,
                               int D) {
    int eid = blockIdx.x;
    int t   = threadIdx.x;            // t = h*D + d
    int h   = t / D;
    int sN  = src[eid], dN = dst[eid];
    float alpha = wbuf[eid * HEADS + h] / s[dN * HEADS + h];
    int HD = HEADS * D;
    atomicAdd(&agg[(size_t)dN * HD + t], alpha * feat[(size_t)sN * HD + t]);
}

// bias + ReLU, and re-emit fp16 features for the next layer's GEMM
__global__ void bias_relu_h16(float* __restrict__ agg, const float* __restrict__ b,
                              _Float16* __restrict__ h16, int HD) {
    int i = blockIdx.x * blockDim.x + threadIdx.x;
    if (i >= NN * HD) return;
    float v = agg[i] + b[i % HD];
    v = v > 0.f ? v : 0.f;
    agg[i] = v;
    h16[i] = (_Float16)v;
}

// final: out[n,c] = mean_h( agg2[n, h*CLS + c] + b2[h*CLS + c] )
__global__ void head_mean(const float* __restrict__ agg, const float* __restrict__ b,
                          float* __restrict__ out) {
    int i = blockIdx.x * blockDim.x + threadIdx.x;      // i = n*CLS + c
    if (i >= NN * CLS) return;
    int c = i % CLS, n = i / CLS;
    float acc = 0.f;
    #pragma unroll
    for (int h = 0; h < HEADS; ++h)
        acc += agg[(size_t)n * (HEADS * CLS) + h * CLS + c] + b[h * CLS + c];
    out[i] = acc * (1.0f / HEADS);
}

// ---------------- host orchestration ----------------
static inline size_t alignUp(size_t x) { return (x + 255) & ~(size_t)255; }

extern "C" void kernel_launch(void* const* d_in, const int* in_sizes, int n_in,
                              void* d_out, int out_size, void* d_ws, size_t ws_size,
                              hipStream_t stream) {
    const float* x   = (const float*)d_in[0];
    const int*   src = (const int*)d_in[1];
    const int*   dst = (const int*)d_in[2];
    const float* W0  = (const float*)d_in[3];
    const float* al0 = (const float*)d_in[4];
    const float* ar0 = (const float*)d_in[5];
    const float* b0  = (const float*)d_in[6];
    const float* W1  = (const float*)d_in[7];
    const float* al1 = (const float*)d_in[8];
    const float* ar1 = (const float*)d_in[9];
    const float* b1  = (const float*)d_in[10];
    const float* W2  = (const float*)d_in[11];
    const float* al2 = (const float*)d_in[12];
    const float* ar2 = (const float*)d_in[13];
    const float* b2  = (const float*)d_in[14];

    // workspace carve-up
    char* ws = (char*)d_ws;
    size_t off = 0;
    _Float16* h16  = (_Float16*)(ws + off); off = alignUp(off + (size_t)NN * 256 * 2);
    _Float16* wt16 = (_Float16*)(ws + off); off = alignUp(off + (size_t)256 * 256 * 2);
    float* feat = (float*)(ws + off); off = alignUp(off + (size_t)NN * 256 * 4);
    float* agg  = (float*)(ws + off); off = alignUp(off + (size_t)NN * 256 * 4);
    float* el   = (float*)(ws + off); off = alignUp(off + (size_t)NN * HEADS * 4);
    float* er   = (float*)(ws + off); off = alignUp(off + (size_t)NN * HEADS * 4);
    float* mmax = (float*)(ws + off); off = alignUp(off + (size_t)NN * HEADS * 4);
    float* ssum = (float*)(ws + off); off = alignUp(off + (size_t)NN * HEADS * 4);
    float* wbuf = (float*)(ws + off); off = alignUp(off + (size_t)EE * HEADS * 4);
    (void)ws_size; (void)n_in; (void)in_sizes; (void)out_size;

    const int TPB = 256;
    auto blocks = [](long n, int tpb) { return (int)((n + tpb - 1) / tpb); };

    // initial fp16 features
    cvt_f16<<<blocks((long)NN * 256, TPB), TPB, 0, stream>>>(x, h16, NN * 256);

    struct Layer { const float *W, *al, *ar, *b; int Fin, Fout, D; bool relu; };
    const Layer layers[3] = {
        { W0, al0, ar0, b0, IN_DIM,        HEADS * HID, HID, true  },
        { W1, al1, ar1, b1, HEADS * HID,   HEADS * HID, HID, true  },
        { W2, al2, ar2, b2, HEADS * HID,   HEADS * CLS, CLS, false },
    };

    for (int l = 0; l < 3; ++l) {
        const Layer& L = layers[l];
        const int HD = L.Fout;

        // weights: transpose + fp16
        cvt_w_t<<<blocks((long)L.Fin * L.Fout, TPB), TPB, 0, stream>>>(
            L.W, wt16, L.Fin, L.Fout);

        // feat = h16 @ W   (WMMA)  — grid: 625 M-tiles of 4 waves x (Fout/16) N-tiles
        dim3 gg(NN / 64, HD / 16);
        gemm_wmma_f16<<<gg, 128, 0, stream>>>(h16, wt16, feat, L.Fin, HD);

        // attention scalar coefficients
        attn_coef<<<blocks((long)NN * HEADS, TPB), TPB, 0, stream>>>(
            feat, L.al, L.ar, el, er, L.D);

        // init reductions
        fill_f32<<<blocks((long)NN * HEADS, TPB), TPB, 0, stream>>>(mmax, -3.0e38f, NN * HEADS);
        fill_f32<<<blocks((long)NN * HEADS, TPB), TPB, 0, stream>>>(ssum, 0.0f, NN * HEADS);
        fill_f32<<<blocks((long)NN * HD, TPB), TPB, 0, stream>>>(agg, 0.0f, NN * HD);

        // edge softmax + aggregate
        edge_max<<<blocks((long)EE * HEADS, TPB), TPB, 0, stream>>>(src, dst, el, er, mmax);
        edge_expsum<<<blocks((long)EE * HEADS, TPB), TPB, 0, stream>>>(
            src, dst, el, er, mmax, wbuf, ssum);
        edge_aggregate<<<EE, HD, 0, stream>>>(src, dst, wbuf, ssum, feat, agg, L.D);

        if (L.relu) {
            bias_relu_h16<<<blocks((long)NN * HD, TPB), TPB, 0, stream>>>(agg, L.b, h16, HD);
        } else {
            head_mean<<<blocks((long)NN * CLS, TPB), TPB, 0, stream>>>(agg, L.b, (float*)d_out);
        }
    }
}
